// JimmyMarkIV_77584289235335
// MI455X (gfx1250) — matrix operand order repaired and checked by reading, not compiled, not run on hardware
//
#include <hip/hip_runtime.h>
#include <hip/hip_bf16.h>
#include <math.h>

// Problem constants (match the reference).
#define N_UNITS   8192
#define INP_UNITS 1024
#define OUT_UNITS 512
#define BATCH     32
#define PROPS     4

typedef __bf16 bf16_t;
typedef __attribute__((ext_vector_type(16))) __bf16 v16bf;
typedef __attribute__((ext_vector_type(8)))  float  v8f;

struct Frag32B { uint4 lo; uint4 hi; };  // 32 bytes == v16bf

// Load 16 bf16 halves as two 16B chunks (lowers to global_load_b128 x2).
__device__ __forceinline__ v16bf load_frag(const bf16_t* p0, const bf16_t* p1) {
  Frag32B t;
  t.lo = *reinterpret_cast<const uint4*>(p0);
  t.hi = *reinterpret_cast<const uint4*>(p1);
  return __builtin_bit_cast(v16bf, t);
}

// ---------------------------------------------------------------------------
// Build dense bf16 M (8192x8192) from packed upper-triangular w.
// M[i][j] = w[pidx(min(i,j), max(i,j))], M[i][i] = 0. One element per thread.
// One-time (per launch) cost; subsequent 128 matvec passes stream M from L2.
// ---------------------------------------------------------------------------
__global__ void build_M(const float* __restrict__ w, bf16_t* __restrict__ M) {
  size_t idx = (size_t)blockIdx.x * blockDim.x + threadIdx.x;
  long long i = (long long)(idx >> 13);        // row  (N = 8192 = 2^13)
  long long j = (long long)(idx & (N_UNITS - 1));
  float v = 0.0f;
  if (i != j) {
    long long a = i < j ? i : j;
    long long b = i < j ? j : i;
    long long widx = a * (long long)N_UNITS - (a * (a + 1)) / 2 + (b - a - 1);
    v = w[widx];
  }
  M[idx] = (bf16_t)v;
}

__global__ void init_state(bf16_t* __restrict__ s0, float* __restrict__ sf) {
  int i = blockIdx.x * blockDim.x + threadIdx.x;
  if (i < N_UNITS) { s0[i] = (bf16_t)0.0f; sf[i] = 0.0f; }
}

__global__ void set_input(const float* __restrict__ x, bf16_t* __restrict__ s) {
  int i = blockIdx.x * blockDim.x + threadIdx.x;
  if (i < INP_UNITS) s[i] = (bf16_t)x[i];
}

__global__ void emit_out(const float* __restrict__ sf, float* __restrict__ out) {
  int i = blockIdx.x * blockDim.x + threadIdx.x;
  if (i < OUT_UNITS) out[i] = sf[N_UNITS - OUT_UNITS + i];
}

// ---------------------------------------------------------------------------
// One propagation step: s_out = tanh(M @ s_in + bias).
// Block = 128 threads (4 waves). Block owns rows [16*blockIdx, 16*blockIdx+16).
// Wave w accumulates K-range [w*2048, (w+1)*2048) with v_wmma_f32_16x16x32_bf16,
// broadcasting the state vector into every B column; column 0 of D is the
// matvec result (lanes 0/16 per the 16x16 f32 C/D layout). LDS reduces the
// 4 wave partials, then tanh + bias, stored as bf16 (next step) and f32 (out).
// ---------------------------------------------------------------------------
__global__ void __launch_bounds__(128)
matvec_tanh(const bf16_t* __restrict__ M, const bf16_t* __restrict__ s_in,
            const float* __restrict__ bias, bf16_t* __restrict__ s_out,
            float* __restrict__ s_out_f32) {
  __shared__ float red[64];

  const int tid  = threadIdx.x;
  const int wave = tid >> 5;
  const int lane = tid & 31;
  const int row0 = blockIdx.x * 16;

  // A-fragment lane mapping (16-bit 16x32 A): lanes 0-15 hold row M=lane,
  // halves 0..7 = K+0..7, halves 8..15 = K+16..23; lanes 16-31 same row,
  // K offsets +8 / +24.
  const int    arow  = row0 + (lane & 15);
  const int    ksel  = (lane >> 4) << 3;             // 0 or 8
  const size_t rbase = (size_t)arow * N_UNITS;

  // B-fragment: lane n (n<16) holds column n, K=0..7,16..23; lane n+16 holds K=8..15,24..31.
  // Broadcast: every column equals the state vector.
  const int bsel = (lane >> 4) << 3;                 // 0 or 8

  v8f acc = {};
  const int kbeg = wave * (N_UNITS / 4);
  const int kend = kbeg + (N_UNITS / 4);
  for (int kk = kbeg; kk < kend; kk += 32) {
    v16bf a = load_frag(M + rbase + kk + ksel, M + rbase + kk + 16 + ksel);
    const bf16_t* sp = s_in + kk + bsel;
    v16bf b = load_frag(sp, sp + 16);
    acc = __builtin_amdgcn_wmma_f32_16x16x32_bf16(
        /*neg_a=*/false, a, /*neg_b=*/false, b,
        /*c_mod=*/(short)0, acc, /*reuse_a=*/false, /*reuse_b=*/false);
  }

  // Column 0 of D: lane 0 holds rows 0..7 (VGPR r -> row r),
  // lane 16 holds rows 8..15.
  if ((lane & 15) == 0) {
    const int base = wave * 16 + (lane >> 4) * 8;
#pragma unroll
    for (int r = 0; r < 8; ++r) red[base + r] = acc[r];
  }
  __syncthreads();

  if (tid < 16) {
    float sum = red[tid] + red[16 + tid] + red[32 + tid] + red[48 + tid];
    float v = tanhf(sum + bias[row0 + tid]);
    s_out[row0 + tid]     = (bf16_t)v;
    s_out_f32[row0 + tid] = v;
  }
}

// ---------------------------------------------------------------------------
// Host side: stream-ordered launches only (graph-capture safe, deterministic).
// d_in order: inputs (32x1024 f32), w (N*(N-1)/2 f32), b (N f32).
// d_out: 32*512 f32.  d_ws: 128 MiB bf16 M + state buffers.
// ---------------------------------------------------------------------------
extern "C" void kernel_launch(void* const* d_in, const int* in_sizes, int n_in,
                              void* d_out, int out_size, void* d_ws, size_t ws_size,
                              hipStream_t stream) {
  (void)in_sizes; (void)n_in; (void)out_size; (void)ws_size;
  const float* inputs = (const float*)d_in[0];
  const float* w      = (const float*)d_in[1];
  const float* bias   = (const float*)d_in[2];
  float*       out    = (float*)d_out;

  char* ws = (char*)d_ws;
  const size_t offM = (size_t)N_UNITS * N_UNITS * sizeof(bf16_t);  // 128 MiB
  bf16_t* M  = (bf16_t*)ws;
  bf16_t* s0 = (bf16_t*)(ws + offM);            // 16 KiB used
  bf16_t* s1 = (bf16_t*)(ws + offM + 32768);    // 16 KiB used
  float*  sf = (float*) (ws + offM + 65536);    // 32 KiB

  // 1) Materialize bf16 M (fits the 192 MB L2 -> later passes are L2-resident).
  {
    const size_t total  = (size_t)N_UNITS * N_UNITS;
    const int    thr    = 256;
    const int    blocks = (int)(total / thr);
    build_M<<<dim3(blocks), dim3(thr), 0, stream>>>(w, M);
  }

  // 2) Zero the carried state.
  init_state<<<(N_UNITS + 255) / 256, 256, 0, stream>>>(s0, sf);

  // 3) 32 batches x 4 props, all serialized on `stream` (state dependency).
  for (int bb = 0; bb < BATCH; ++bb) {
    set_input<<<(INP_UNITS + 255) / 256, 256, 0, stream>>>(inputs + bb * INP_UNITS, s0);
    bf16_t* cur = s0;
    bf16_t* nxt = s1;
    for (int p = 0; p < PROPS; ++p) {
      matvec_tanh<<<N_UNITS / 16, 128, 0, stream>>>(M, cur, bias, nxt, sf);
      bf16_t* t = cur; cur = nxt; nxt = t;  // PROPS even -> state ends in s0
    }
    emit_out<<<(OUT_UNITS + 255) / 256, 256, 0, stream>>>(sf, out + bb * OUT_UNITS);
  }
}
